// gqattn_10488310136954
// MI455X (gfx1250) — matrix operand order repaired and checked by reading, not compiled, not run on hardware
//
#include <hip/hip_runtime.h>

// ---------------------------------------------------------------------------
// GQA attention block for MI455X (gfx1250, wave32).
// - All matmuls on v_wmma_f32_16x16x32_bf16 (bf16 in, f32 accum).
// - GEMM tiles streamed with global_load_async_to_lds_b128 (ASYNCcnt),
//   double-buffered LDS.
// - Flash-attention K/V tiles streamed by the Tensor Data Mover
//   (tensor_load_to_lds + s_wait_tensorcnt), double-buffered.
// ---------------------------------------------------------------------------

typedef __attribute__((ext_vector_type(16))) __bf16 v16bf;
typedef __attribute__((ext_vector_type(8)))  float  v8f;
typedef unsigned int v4u  __attribute__((ext_vector_type(4)));
typedef int          v8i_ __attribute__((ext_vector_type(8)));
typedef int          v4i_ __attribute__((ext_vector_type(4)));

#define DEV __device__ __forceinline__

constexpr int cB   = 2;
constexpr int cS   = 2048;
constexpr int cD   = 2048;
constexpr int cNH  = 32;
constexpr int cNKV = 8;
constexpr int cHD  = 64;
constexpr int cRATIO = cNH / cNKV;   // 4
constexpr int cM = cB * cS;          // 4096

// --- fragment loaders (wave32 WMMA register layouts) -----------------------
DEV v16bf ld_frag2(const __bf16* p0, const __bf16* p1) {
    union { v16bf v; uint4 q[2]; } u;
    u.q[0] = *(const uint4*)p0;
    u.q[1] = *(const uint4*)p1;
    return u.v;
}
DEV v16bf ld_frag_c(const __bf16* p) {
    union { v16bf v; uint4 q[2]; } u;
    u.q[0] = *(const uint4*)p;
    u.q[1] = *(const uint4*)(p + 8);
    return u.v;
}
DEV v8f wmma_bf16(v16bf a, v16bf b, v8f c) {
    return __builtin_amdgcn_wmma_f32_16x16x32_bf16(false, a, false, b,
                                                   (short)0, c, false, false);
}

// --- LDS helpers -----------------------------------------------------------
DEV unsigned lds_off(const void* p) {            // generic ptr -> LDS byte offset
    return (unsigned)(uintptr_t)p;
}
DEV void async_ld_b128(unsigned lds, const void* g) {
    asm volatile("global_load_async_to_lds_b128 %0, %1, off"
                 :: "v"(lds), "v"(g) : "memory");
}

// --- TDM: 2D tile load (data_size = 2 bytes) -------------------------------
DEV void tdm_load_2d(unsigned lds, const void* gaddr,
                     unsigned tens_d0, unsigned tens_d1, unsigned stride0,
                     unsigned tile_d0, unsigned tile_d1) {
    unsigned long long ga = (unsigned long long)gaddr;
    v4u g0;
    g0[0] = 1u;                                          // count=1 (valid), user mode
    g0[1] = lds;                                         // lds_addr
    g0[2] = (unsigned)ga;                                // global_addr[31:0]
    g0[3] = (unsigned)((ga >> 32) & 0x1FFFFFFull)        // global_addr[56:32]
            | 0x80000000u;                               // type = 2 ("image")
    v8i_ g1;
    g1[0] = (int)(1u << 16);                             // data_size = 1 -> 2 bytes
    g1[1] = (int)((tens_d0 & 0xFFFFu) << 16);            // tensor_dim0[15:0]
    g1[2] = (int)(((tens_d0 >> 16) & 0xFFFFu) |
                  ((tens_d1 & 0xFFFFu) << 16));          // dim0 hi | dim1 lo
    g1[3] = (int)(((tens_d1 >> 16) & 0xFFFFu) |
                  ((tile_d0 & 0xFFFFu) << 16));          // dim1 hi | tile_dim0
    g1[4] = (int)(tile_d1 & 0xFFFFu);                    // tile_dim1 (tile_dim2=0)
    g1[5] = (int)stride0;                                // tensor_dim0_stride lo
    g1[6] = 0;
    g1[7] = 0;
    v4i_ z4 = {0, 0, 0, 0};
    v8i_ z8 = {0, 0, 0, 0, 0, 0, 0, 0};
    __builtin_amdgcn_tensor_load_to_lds(g0, g1, z4, z4, z8, 0);
}

// --- kernel 1a: fp32 -> bf16 (x) -------------------------------------------
__global__ __launch_bounds__(256) void k_cvt_bf16(const float* __restrict__ in,
                                                  __bf16* __restrict__ out, int n) {
    int i = blockIdx.x * 256 + threadIdx.x;
    if (i < n) out[i] = (__bf16)in[i];
}

// --- kernel 1b: fp32 W[K,N] -> bf16 WT[N,K] (LDS-tiled transpose) ----------
__global__ __launch_bounds__(256)
void k_cvt_T(const float* __restrict__ in, __bf16* __restrict__ out, int N, int K) {
    __shared__ float tle[64][65];
    const int t = threadIdx.x;
    const int k0 = blockIdx.x * 64, n0 = blockIdx.y * 64;
    #pragma unroll
    for (int r = 0; r < 16; ++r) {
        int row = (t >> 6) + r * 4, col = t & 63;
        tle[row][col] = in[(size_t)(k0 + row) * N + n0 + col];
    }
    __syncthreads();
    #pragma unroll
    for (int r = 0; r < 16; ++r) {
        int row = (t >> 6) + r * 4, col = t & 63;
        out[(size_t)(n0 + row) * K + k0 + col] = (__bf16)tle[col][row];
    }
}

// --- kernel 2: bf16 GEMM  C[M,N] = A[M,K] @ WT[N,K]^T ----------------------
// 256 threads / 8 waves, block tile 128x128, K-step 32, double-buffered LDS
// filled by async-to-LDS loads. Wave (wm, wn) owns 32(M) x 64(N): 8 WMMAs/step.
__global__ __launch_bounds__(256)
void k_gemm_bf16(const __bf16* __restrict__ A, const __bf16* __restrict__ WT,
                 float* __restrict__ C, int M, int N, int K) {
    __shared__ __bf16 at[2][128 * 32];   // [row][k]
    __shared__ __bf16 wt[2][128 * 32];   // [n][k]
    const int t = threadIdx.x;
    const int lane = t & 31, wave = t >> 5;
    const int hl = lane >> 4, ln = lane & 15;
    const int wm = wave & 3, wn = wave >> 2;
    const int m0 = blockIdx.y * 128;
    const int n0 = blockIdx.x * 128;
    const int row = t >> 1, seg = (t & 1) * 16;          // copy mapping (16 elem)

    const __bf16* gA = A  + (size_t)(m0 + row) * K + seg;
    const __bf16* gW = WT + (size_t)(n0 + row) * K + seg;

    v8f zero = {0.f,0.f,0.f,0.f,0.f,0.f,0.f,0.f};
    v8f acc[2][4] = {{zero,zero,zero,zero},{zero,zero,zero,zero}};

    const int steps = K / 32;
    auto issue = [&](int i, int buf) {
        unsigned la = lds_off(&at[buf][row * 32 + seg]);
        unsigned lw = lds_off(&wt[buf][row * 32 + seg]);
        const __bf16* ga = gA + i * 32;
        const __bf16* gw = gW + i * 32;
        async_ld_b128(la,      ga);
        async_ld_b128(la + 16, ga + 8);
        async_ld_b128(lw,      gw);
        async_ld_b128(lw + 16, gw + 8);
    };
    issue(0, 0);
    for (int i = 0; i < steps; ++i) {
        __syncthreads();                         // prev compute done: alt buffer free
        if (i + 1 < steps) {
            issue(i + 1, (i + 1) & 1);
            asm volatile("s_wait_asynccnt 0x4" ::: "memory");   // tile i landed
        } else {
            asm volatile("s_wait_asynccnt 0x0" ::: "memory");
        }
        __syncthreads();                         // tile i visible to all waves
        const int buf = i & 1;
        v16bf af[2], bf[4];
        #pragma unroll
        for (int a = 0; a < 2; ++a) {
            const __bf16* ar = &at[buf][(wm * 32 + a * 16 + ln) * 32];
            af[a] = ld_frag2(ar + hl * 16, ar + hl * 16 + 8);
        }
        #pragma unroll
        for (int b = 0; b < 4; ++b)
            bf[b] = ld_frag_c(&wt[buf][(wn * 64 + b * 16 + ln) * 32 + hl * 16]);
        #pragma unroll
        for (int a = 0; a < 2; ++a)
            #pragma unroll
            for (int b = 0; b < 4; ++b)
                acc[a][b] = wmma_bf16(af[a], bf[b], acc[a][b]);
    }
    #pragma unroll
    for (int a = 0; a < 2; ++a)
        #pragma unroll
        for (int b = 0; b < 4; ++b)
            #pragma unroll
            for (int r = 0; r < 8; ++r) {
                int rr = m0 + wm * 32 + a * 16 + hl * 8 + r;
                int cc = n0 + wn * 64 + b * 16 + ln;
                C[(size_t)rr * N + cc] = acc[a][b][r];
            }
}

// --- kernel 3: per-head RMSNorm + RoPE, fp32 -> bf16 head-major ------------
__global__ __launch_bounds__(256)
void k_normrope(const float* __restrict__ raw, const float* __restrict__ scale,
                const float* __restrict__ cosb, const float* __restrict__ sinb,
                __bf16* __restrict__ out, int H) {
    const int t = threadIdx.x, lane = t & 31, wave = t >> 5;
    long g  = (long)blockIdx.x * 8 + wave;
    int  h  = (int)(g % H);
    long bs = g / H;
    int  s  = (int)(bs % cS);
    long b  = bs / cS;
    const float* rp = raw + bs * (long)(H * 64) + h * 64;
    float x1 = rp[lane], x2 = rp[lane + 32];
    float ss = x1 * x1 + x2 * x2;
    #pragma unroll
    for (int off = 16; off; off >>= 1) ss += __shfl_xor(ss, off, 32);
    float r  = rsqrtf(ss * (1.f / 64.f) + 1e-6f);
    float n1 = x1 * r * scale[lane];
    float n2 = x2 * r * scale[lane + 32];
    float o1 = n1 * cosb[s * 64 + lane]      - n2 * sinb[s * 64 + lane];
    float o2 = n2 * cosb[s * 64 + lane + 32] + n1 * sinb[s * 64 + lane + 32];
    size_t ob = (((size_t)b * H + h) * cS + s) * 64;
    out[ob + lane]      = (__bf16)o1;
    out[ob + lane + 32] = (__bf16)o2;
}

// --- kernel 3b: V fp32 [b,s,kh,hd] -> bf16 [b,kh,hd,s] ---------------------
__global__ __launch_bounds__(256)
void k_vtrans(const float* __restrict__ raw, __bf16* __restrict__ out) {
    const int t = threadIdx.x, lane = t & 31, wave = t >> 5;
    long g  = (long)blockIdx.x * 8 + wave;
    int  h  = (int)(g % cNKV);
    long bs = g / cNKV;
    int  s  = (int)(bs % cS);
    long b  = bs / cS;
    const float* rp = raw + bs * (long)(cNKV * 64) + h * 64;
    size_t base = ((size_t)b * cNKV + h) * 64;
    out[(base + lane)      * cS + s] = (__bf16)rp[lane];
    out[(base + lane + 32) * cS + s] = (__bf16)rp[lane + 32];
}

// --- kernel 4: causal flash attention, TDM-fed, double-buffered ------------
// Grid (S/128, NH, B); block 256 = 8 waves, each wave owns 16 query rows.
__global__ __launch_bounds__(256)
void k_flash(const __bf16* __restrict__ Qh, const __bf16* __restrict__ Kh,
             const __bf16* __restrict__ Vt, __bf16* __restrict__ O) {
    __shared__ __bf16 kt_s[2][64 * 64];      // [key][hd]
    __shared__ __bf16 vt_s[2][64 * 64];      // [hd][key]
    __shared__ __bf16 pt_s[8 * 16 * 64];     // per-wave P tile [m][key]
    const int t = threadIdx.x, lane = t & 31, wave = t >> 5;
    const int hl = lane >> 4, ln = lane & 15;
    const int qt = blockIdx.x, h = blockIdx.y, b = blockIdx.z;
    const int kh = h / cRATIO;
    const int q0 = qt * 128 + wave * 16;

    const __bf16* Qb = Qh + ((size_t)b * cNH  + h ) * cS * 64;
    const __bf16* Kb = Kh + ((size_t)b * cNKV + kh) * cS * 64;
    const __bf16* Vb = Vt + ((size_t)b * cNKV + kh) * 64 * (size_t)cS;

    v16bf qf[2];
    {
        const __bf16* qr = Qb + (size_t)(q0 + ln) * 64;
        qf[0] = ld_frag2(qr +      hl * 16, qr +  8 + hl * 16);
        qf[1] = ld_frag2(qr + 32 + hl * 16, qr + 40 + hl * 16);
    }
    v8f zero = {0.f,0.f,0.f,0.f,0.f,0.f,0.f,0.f};
    v8f oacc[4] = {zero, zero, zero, zero};
    float mi[8], li[8];
    #pragma unroll
    for (int r = 0; r < 8; ++r) { mi[r] = -1e30f; li[r] = 0.f; }

    const int nkt = 2 * qt + 2;
    auto issue = [&](int kt, int buf) {      // executed by wave 0 only
        // K tile: rows = 64 keys (stride 64 elem), cols = 64 hd (contiguous)
        tdm_load_2d(lds_off(&kt_s[buf][0]), Kb + (size_t)kt * 64 * 64,
                    64, cS, 64, 64, 64);
        // V^T tile: rows = 64 hd (stride S elem), cols = 64 keys (contiguous)
        tdm_load_2d(lds_off(&vt_s[buf][0]), Vb + (size_t)kt * 64,
                    cS, 64, cS, 64, 64);
    };
    if (wave == 0) issue(0, 0);

    for (int kt = 0; kt < nkt; ++kt) {
        const int kb = kt * 64;
        __syncthreads();                     // alt buffers free for refill
        if (wave == 0) {
            if (kt + 1 < nkt) {
                issue(kt + 1, (kt + 1) & 1);
                __builtin_amdgcn_s_wait_tensorcnt(2);   // tile kt complete
            } else {
                __builtin_amdgcn_s_wait_tensorcnt(0);
            }
        }
        __syncthreads();                     // tile kt visible block-wide
        const __bf16* ktile = kt_s[kt & 1];
        const __bf16* vtile = vt_s[kt & 1];

        // S = Q @ K^T : 8 WMMAs
        v8f sf[4];
        #pragma unroll
        for (int nt = 0; nt < 4; ++nt) {
            v8f a = zero;
            #pragma unroll
            for (int kk = 0; kk < 2; ++kk) {
                v16bf bfrg = ld_frag_c(&ktile[(nt * 16 + ln) * 64 + kk * 32 + hl * 16]);
                a = wmma_bf16(qf[kk], bfrg, a);
            }
            sf[nt] = a;
        }

        // scale 1/sqrt(64), causal mask, row max
        float rmax[8];
        #pragma unroll
        for (int r = 0; r < 8; ++r) rmax[r] = -1e30f;
        #pragma unroll
        for (int nt = 0; nt < 4; ++nt) {
            int keyg = kb + nt * 16 + ln;
            #pragma unroll
            for (int r = 0; r < 8; ++r) {
                int qg = q0 + hl * 8 + r;
                float v = sf[nt][r] * 0.125f;
                if (keyg > qg) v = -1e30f;
                sf[nt][r] = v;
                rmax[r] = fmaxf(rmax[r], v);
            }
        }
        #pragma unroll
        for (int r = 0; r < 8; ++r) {
            #pragma unroll
            for (int off = 8; off; off >>= 1)
                rmax[r] = fmaxf(rmax[r], __shfl_xor(rmax[r], off, 32));
            float nm = fmaxf(mi[r], rmax[r]);
            float sc = __expf(mi[r] - nm);
            mi[r] = nm;
            li[r] *= sc;
            #pragma unroll
            for (int nt = 0; nt < 4; ++nt) oacc[nt][r] *= sc;
        }
        // P = exp(S - m), row sums
        float rs[8];
        #pragma unroll
        for (int r = 0; r < 8; ++r) rs[r] = 0.f;
        #pragma unroll
        for (int nt = 0; nt < 4; ++nt)
            #pragma unroll
            for (int r = 0; r < 8; ++r) {
                float p = __expf(sf[nt][r] - mi[r]);
                sf[nt][r] = p;
                rs[r] += p;
            }
        #pragma unroll
        for (int r = 0; r < 8; ++r) {
            #pragma unroll
            for (int off = 8; off; off >>= 1) rs[r] += __shfl_xor(rs[r], off, 32);
            li[r] += rs[r];
        }

        // C-layout -> A-layout transpose of P through wave-private LDS
        __bf16* pw = &pt_s[wave * 1024];
        #pragma unroll
        for (int nt = 0; nt < 4; ++nt)
            #pragma unroll
            for (int r = 0; r < 8; ++r)
                pw[(hl * 8 + r) * 64 + nt * 16 + ln] = (__bf16)sf[nt][r];
        asm volatile("s_wait_dscnt 0" ::: "memory");
        v16bf pf[2];
        {
            const __bf16* pr = pw + ln * 64;
            pf[0] = ld_frag2(pr +      hl * 16, pr +  8 + hl * 16);
            pf[1] = ld_frag2(pr + 32 + hl * 16, pr + 40 + hl * 16);
        }
        // O += P @ V : 8 WMMAs
        #pragma unroll
        for (int nt = 0; nt < 4; ++nt)
            #pragma unroll
            for (int kk = 0; kk < 2; ++kk) {
                v16bf bfrg = ld_frag_c(&vtile[(nt * 16 + ln) * 64 + kk * 32 + hl * 16]);
                oacc[nt] = wmma_bf16(pf[kk], bfrg, oacc[nt]);
            }
    }

    #pragma unroll
    for (int r = 0; r < 8; ++r) {
        float inv = 1.f / li[r];
        int s = q0 + hl * 8 + r;
        size_t ob = ((size_t)b * cS + s) * (size_t)(cNH * cHD) + h * 64;
        #pragma unroll
        for (int nt = 0; nt < 4; ++nt)
            O[ob + nt * 16 + ln] = (__bf16)(oacc[nt][r] * inv);
    }
}

// ---------------------------------------------------------------------------
extern "C" void kernel_launch(void* const* d_in, const int* in_sizes, int n_in,
                              void* d_out, int out_size, void* d_ws, size_t ws_size,
                              hipStream_t stream) {
    (void)in_sizes; (void)n_in; (void)out_size; (void)ws_size;
    const float* x    = (const float*)d_in[0];
    const float* wq   = (const float*)d_in[1];
    const float* wk   = (const float*)d_in[2];
    const float* wv   = (const float*)d_in[3];
    const float* wo   = (const float*)d_in[4];
    const float* qsc  = (const float*)d_in[5];
    const float* ksc  = (const float*)d_in[6];
    const float* cosb = (const float*)d_in[7];
    const float* sinb = (const float*)d_in[8];
    // d_in[9] (bool mask) unused: causal mask applied analytically.

    char* ws = (char*)d_ws;
    const size_t MB = 1ull << 20;
    __bf16* xb   = (__bf16*)(ws +   0 * MB);   // 16 MB  x bf16
    __bf16* wqT  = (__bf16*)(ws +  16 * MB);   //  8 MB  [N=2048][K=2048]
    __bf16* wkT  = (__bf16*)(ws +  24 * MB);   //  2 MB  [512][2048]
    __bf16* wvT  = (__bf16*)(ws +  26 * MB);   //  2 MB
    __bf16* woT  = (__bf16*)(ws +  28 * MB);   //  8 MB  [2048][2048]
    float*  qraw = (float*) (ws +  36 * MB);   // 32 MB  (region reused by Ob)
    float*  kraw = (float*) (ws +  68 * MB);   //  8 MB
    float*  vraw = (float*) (ws +  76 * MB);   //  8 MB
    __bf16* Qh   = (__bf16*)(ws +  84 * MB);   // 16 MB  [b,h,s,hd]
    __bf16* Kh   = (__bf16*)(ws + 100 * MB);   //  4 MB  [b,kh,s,hd]
    __bf16* Vtb  = (__bf16*)(ws + 104 * MB);   //  4 MB  [b,kh,hd,s]
    __bf16* Ob   = (__bf16*)(ws +  36 * MB);   // 16 MB  attn out (reuses qraw)

    // 1. fp32 -> bf16 (x) and fp32 -> bf16 transposed weights
    k_cvt_bf16<<<(cM * cD) / 256, 256, 0, stream>>>(x, xb, cM * cD);
    k_cvt_T<<<dim3(cD / 64, (cNH * cHD) / 64), 256, 0, stream>>>(wq, wqT, cNH * cHD, cD);
    k_cvt_T<<<dim3(cD / 64, (cNKV * cHD) / 64), 256, 0, stream>>>(wk, wkT, cNKV * cHD, cD);
    k_cvt_T<<<dim3(cD / 64, (cNKV * cHD) / 64), 256, 0, stream>>>(wv, wvT, cNKV * cHD, cD);
    k_cvt_T<<<dim3((cNH * cHD) / 64, cD / 64), 256, 0, stream>>>(wo, woT, cD, cNH * cHD);

    // 2. QKV projections (async-LDS double-buffered WMMA GEMM)
    k_gemm_bf16<<<dim3((cNH * cHD) / 128, cM / 128), 256, 0, stream>>>(xb, wqT, qraw, cM, cNH * cHD, cD);
    k_gemm_bf16<<<dim3((cNKV * cHD) / 128, cM / 128), 256, 0, stream>>>(xb, wkT, kraw, cM, cNKV * cHD, cD);
    k_gemm_bf16<<<dim3((cNKV * cHD) / 128, cM / 128), 256, 0, stream>>>(xb, wvT, vraw, cM, cNKV * cHD, cD);

    // 3. RMSNorm + RoPE (Q, K) and V transpose -> bf16 head-major layouts
    k_normrope<<<(cB * cS * cNH) / 8, 256, 0, stream>>>(qraw, qsc, cosb, sinb, Qh, cNH);
    k_normrope<<<(cB * cS * cNKV) / 8, 256, 0, stream>>>(kraw, ksc, cosb, sinb, Kh, cNKV);
    k_vtrans  <<<(cB * cS * cNKV) / 8, 256, 0, stream>>>(vraw, Vtb);

    // 4. causal flash attention (TDM-fed)
    k_flash<<<dim3(cS / 128, cNH, cB), 256, 0, stream>>>(Qh, Kh, Vtb, Ob);

    // 5. output projection -> d_out (fp32)
    k_gemm_bf16<<<dim3(cD / 128, cM / 128), 256, 0, stream>>>(Ob, woT, (float*)d_out, cM, cD, cNH * cHD);
}
